// DigitConvolutionalModel_16183436771772
// MI455X (gfx1250) — compile-verified, matrix operand order checked by reading, not run on hardware
//
#include <hip/hip_runtime.h>
#include <hip/hip_bf16.h>

typedef __attribute__((ext_vector_type(16))) _Float16 v16h;
typedef __attribute__((ext_vector_type(4)))  _Float16 v4h;
typedef __attribute__((ext_vector_type(8)))  float    v8f;
typedef __attribute__((ext_vector_type(2)))  _Float16 v2h;

#define B_TOTAL 32768
#define IMG     28
#define OUT_HW  26
#define FLAT    676
#define HALF_FLAT 338
#define HID     300
#define NCLS    10

// Padded GEMM geometry (16x16x32 f16 WMMA tiles)
#define K1T 22          // K padded 676 -> 704
#define K2T 10          // K padded 300 -> 320
#define NT_H 20         // hidden N padded to 320
#define NHALF 10        // N processed in two passes of 10 tiles (80 acc VGPRs)
#define NT_O 1

#define TILE_H 512                  // halves per packed fragment tile (1024 B)
#define SLICE  (NT_H * TILE_H)      // full k-slice of W1/W2 (20 tiles)
#define HSLICE (NHALF * TILE_H)     // staged half-slice (10 tiles, 10 KiB)

// Workspace layout (units: _Float16 halves)
#define W1P_OFF 0
#define W2P_OFF (W1P_OFF + K1T * SLICE)
#define W3P_OFF (W2P_OFF + K2T * SLICE)
#define H0P_OFF (W3P_OFF + K2T * NT_O * TILE_H)

// Tile storage layout (bank-conflict-free for wave32 LDS reads):
//   halves index within tile = chunk*128 + lane*4 + (half&3),  chunk = half>>2
// A lane's v16h fragment = 4 x ds_load_b64 with 8B lane stride -> all 64 banks hit once.

__device__ __forceinline__ int kmap(int hi, int half) {
  int j = half >> 1, p = half & 1;
  return ((j >= 4) ? 16 : 0) + 8 * hi + 2 * (j & 3) + p;
}
__device__ __forceinline__ void kinv(int kk, int r, int& lane, int& half) {
  int j = 0;
  if (kk >= 16) { j = 4; kk -= 16; }
  int hi = kk >> 3, rem = kk & 7;
  half = 2 * (j + (rem >> 1)) + (rem & 1);
  lane = hi * 16 + r;
}

union FragU { v16h v; v4h q[4]; };

__device__ __forceinline__ v16h frag_load(const _Float16* tile, int lane) {
  FragU u;
#pragma unroll
  for (int c = 0; c < 4; ++c)
    u.q[c] = *(const v4h*)(tile + c * 128 + lane * 4);   // b64, conflict-free in LDS
  return u.v;
}

// cooperative 10KiB half-slice staging: 256 threads x 5 x b64 (10 VGPRs only)
__device__ __forceinline__ void hslice_load(const _Float16* __restrict__ g, int tid, uint2 t[5]) {
#pragma unroll
  for (int c = 0; c < 5; ++c)
    t[c] = *(const uint2*)(g + c * 1024 + tid * 4);
}
__device__ __forceinline__ void hslice_store(_Float16* l, int tid, const uint2 t[5]) {
#pragma unroll
  for (int c = 0; c < 5; ++c)
    *(uint2*)(l + c * 1024 + tid * 4) = t[c];
}

// ---------------- weight repack: f32 row-major (K,N) -> f16 WMMA B fragments ----------------
__global__ void pack_w(const float* __restrict__ W, _Float16* __restrict__ dst,
                       int K, int N, int ntiles, int total) {
  int tid = blockIdx.x * blockDim.x + threadIdx.x;
  if (tid >= total) return;
  int i    = tid & 3;
  int lane = (tid >> 2) & 31;
  int c    = (tid >> 7) & 3;
  int tile = tid >> 9;
  int half = c * 4 + i;
  int kt = tile / ntiles, nt = tile - kt * ntiles;
  int k = kt * 32 + kmap(lane >> 4, half);
  int n = nt * 16 + (lane & 15);
  _Float16 v = (_Float16)0.0f;
  if (k < K && n < N) v = (_Float16)W[(size_t)k * N + n];
  dst[tid] = v;
}

// ------- conv 3x3 valid; 2 adjacent outputs per thread -> single b32 packed store -------
__global__ void conv_pack(const float* __restrict__ x, const float* __restrict__ cw,
                          _Float16* __restrict__ h0p) {
  int idx = blockIdx.x * blockDim.x + threadIdx.x;
  if (idx >= B_TOTAL * HALF_FLAT) return;
  int b = idx / HALF_FLAT, pp = idx - b * HALF_FLAT;
  int f0 = pp * 2;
  const float* xb = x + (size_t)b * (IMG * IMG);
  float w[9];
#pragma unroll
  for (int t = 0; t < 9; ++t) w[t] = cw[t];
  float s[2];
#pragma unroll
  for (int q = 0; q < 2; ++q) {
    int f = f0 + q;
    int oy = f / OUT_HW, ox = f - oy * OUT_HW;
    float a = 0.f;
#pragma unroll
    for (int i = 0; i < 3; ++i)
#pragma unroll
      for (int j = 0; j < 3; ++j)
        a += xb[(oy + i) * IMG + ox + j] * w[i * 3 + j];
    s[q] = a;
  }
  int mt = b >> 4, r = b & 15;
  int kt = f0 >> 5, kk = f0 & 31;
  int lane, half;
  kinv(kk, r, lane, half);
  v2h pr = {(_Float16)s[0], (_Float16)s[1]};
  size_t base = ((size_t)mt * K1T + kt) * TILE_H;
  *(v2h*)(h0p + base + (half >> 2) * 128 + lane * 4 + (half & 3)) = pr;
}

// One GEMM pass over 10 N-tiles with double-buffered LDS W slices.
// AG=true: A fragments streamed from global (with next-A prefetch); else from LDS.
template <bool AG>
__device__ __forceinline__ void gemm_pass(
    int KT,
    const _Float16* __restrict__ wbase,   // half-slice base; + kt*SLICE per k-step
    const _Float16* __restrict__ aG,      // global A base (tile kt at + kt*TILE_H)
    const _Float16* __restrict__ aL,      // LDS A base
    _Float16* sW, int tid, int lane, v8f* acc) {
  {
    uint2 t0[5];
    hslice_load(wbase, tid, t0);
    hslice_store(sW, tid, t0);
  }
  __syncthreads();
  v16h a_cur = {};
  if (AG) a_cur = frag_load(aG, lane);
  for (int kt = 0; kt < KT; ++kt) {
    const int cur = kt & 1;
    const bool more = (kt + 1 < KT);
    uint2 t[5];
    v16h a_nxt;
    if (more) hslice_load(wbase + (size_t)(kt + 1) * SLICE, tid, t);
    if (AG) {
      if (more) a_nxt = frag_load(aG + (kt + 1) * TILE_H, lane);
    } else {
      a_cur = frag_load(aL + kt * TILE_H, lane);
    }
    const _Float16* bb = sW + cur * HSLICE;
    v16h f0 = frag_load(bb, lane);                 // distance-2 DS pipeline
    v16h f1 = frag_load(bb + TILE_H, lane);
#pragma unroll
    for (int nt = 0; nt < NHALF; ++nt) {
      v16h fn = f1;
      if (nt + 2 < NHALF) fn = frag_load(bb + (nt + 2) * TILE_H, lane);
      acc[nt] = __builtin_amdgcn_wmma_f32_16x16x32_f16(false, a_cur, false, f0,
                                                       (short)0, acc[nt], false, false);
      f0 = f1; f1 = fn;
      if (nt == 5 && more)                         // globals landed behind ~6 WMMAs
        hslice_store(sW + (1 - cur) * HSLICE, tid, t);
    }
    if (AG && more) a_cur = a_nxt;
    __syncthreads();
  }
}

// bias + relu + f16-convert + scatter into next-layer A-fragment layout in LDS
__device__ __forceinline__ void epilogue(const v8f* acc, int ntbase,
                                         const float* __restrict__ bvec,
                                         _Float16* dstH, int n, int hi) {
#pragma unroll
  for (int j = 0; j < NHALF; ++j) {
    int c = (ntbase + j) * 16 + n;
    float bias = (c < HID) ? bvec[c] : 0.f;
    int kt2 = c >> 5, kk = c & 31;
#pragma unroll
    for (int v = 0; v < 8; ++v) {
      float t = fmaxf(acc[j][v] + bias, 0.f);
      int lane2, half2;
      kinv(kk, v + 8 * hi, lane2, half2);
      dstH[kt2 * TILE_H + (half2 >> 2) * 128 + lane2 * 4 + (half2 & 3)] = (_Float16)t;
    }
  }
}

// ---------------- fused 3-layer MLP, one wave per 16-row batch tile ----------------
__global__ __launch_bounds__(256, 1) void fused_mlp(
    const _Float16* __restrict__ ws,
    const float* __restrict__ b1, const float* __restrict__ b2,
    const float* __restrict__ b3, float* __restrict__ out) {
  extern __shared__ _Float16 smem[];
  _Float16* sW  = smem;                            // 2 * HSLICE (20 KiB W double buffer)
  _Float16* sHA = smem + 2 * HSLICE;               // 8 waves * 5120 halves (h1, 80 KiB)
  _Float16* sHB = sHA + 8 * (K2T * TILE_H);        // 8 waves * 5120 halves (h2, 80 KiB)

  const int tid  = threadIdx.x;
  const int wv   = tid >> 5;
  const int lane = tid & 31;
  const int n    = lane & 15;
  const int hi   = lane >> 4;
  const int mtile = blockIdx.x * 8 + wv;

  const _Float16* w1p = ws + W1P_OFF;
  const _Float16* w2p = ws + W2P_OFF;
  const _Float16* w3p = ws + W3P_OFF;
  const _Float16* h0p = ws + H0P_OFF;
  _Float16* myHA = sHA + wv * (K2T * TILE_H);
  _Float16* myHB = sHB + wv * (K2T * TILE_H);
  const _Float16* aBase = h0p + (size_t)mtile * K1T * TILE_H;

  const v8f vzero = {0.f, 0.f, 0.f, 0.f, 0.f, 0.f, 0.f, 0.f};
  v8f acc[NHALF];

  // ========== layer 1: h0(16x704) @ W1(704x320), two N-halves ==========
#pragma unroll
  for (int t = 0; t < NHALF; ++t) acc[t] = vzero;
  gemm_pass<true>(K1T, w1p, aBase, nullptr, sW, tid, lane, acc);
  epilogue(acc, 0, b1, myHA, n, hi);

#pragma unroll
  for (int t = 0; t < NHALF; ++t) acc[t] = vzero;
  gemm_pass<true>(K1T, w1p + HSLICE, aBase, nullptr, sW, tid, lane, acc);
  epilogue(acc, NHALF, b1, myHA, n, hi);

  // ========== layer 2: h1(16x320) @ W2(320x320), two N-halves ==========
#pragma unroll
  for (int t = 0; t < NHALF; ++t) acc[t] = vzero;
  gemm_pass<false>(K2T, w2p, nullptr, myHA, sW, tid, lane, acc);
  epilogue(acc, 0, b2, myHB, n, hi);

#pragma unroll
  for (int t = 0; t < NHALF; ++t) acc[t] = vzero;
  gemm_pass<false>(K2T, w2p + HSLICE, nullptr, myHA, sW, tid, lane, acc);
  epilogue(acc, NHALF, b2, myHB, n, hi);

  // ========== layer 3: h2(16x320) @ W3(320x16) ==========
  {
    uint2 t0[5];                                   // W3 is exactly one 10 KiB half-slice
    hslice_load(w3p, tid, t0);
    hslice_store(sW, tid, t0);
  }
  __syncthreads();
  v8f acc3 = vzero;
  for (int kt = 0; kt < K2T; ++kt) {
    v16h a  = frag_load(myHB + kt * TILE_H, lane);
    v16h bf = frag_load(sW + kt * TILE_H, lane);
    acc3 = __builtin_amdgcn_wmma_f32_16x16x32_f16(false, a, false, bf,
                                                  (short)0, acc3, false, false);
  }
  if (n < NCLS) {
    float bias = b3[n];
    int rowbase = mtile * 16 + 8 * hi;
#pragma unroll
    for (int v = 0; v < 8; ++v)
      out[(size_t)(rowbase + v) * NCLS + n] = acc3[v] + bias;
  }
}

extern "C" void kernel_launch(void* const* d_in, const int* in_sizes, int n_in,
                              void* d_out, int out_size, void* d_ws, size_t ws_size,
                              hipStream_t stream) {
  const float* x  = (const float*)d_in[0];
  const float* cw = (const float*)d_in[1];
  const float* W1 = (const float*)d_in[2];
  const float* b1 = (const float*)d_in[3];
  const float* W2 = (const float*)d_in[4];
  const float* b2 = (const float*)d_in[5];
  const float* W3 = (const float*)d_in[6];
  const float* b3 = (const float*)d_in[7];
  float* out = (float*)d_out;
  _Float16* ws = (_Float16*)d_ws;   // ~44.7 MiB used

  int t1 = K1T * NT_H * TILE_H;
  pack_w<<<(t1 + 255) / 256, 256, 0, stream>>>(W1, ws + W1P_OFF, FLAT, HID, NT_H, t1);
  int t2 = K2T * NT_H * TILE_H;
  pack_w<<<(t2 + 255) / 256, 256, 0, stream>>>(W2, ws + W2P_OFF, HID, HID, NT_H, t2);
  int t3 = K2T * NT_O * TILE_H;
  pack_w<<<(t3 + 255) / 256, 256, 0, stream>>>(W3, ws + W3P_OFF, HID, NCLS, NT_O, t3);

  int ct = B_TOTAL * HALF_FLAT;
  conv_pack<<<(ct + 255) / 256, 256, 0, stream>>>(x, cw, ws + H0P_OFF);

  size_t lds = (size_t)(2 * HSLICE + 16 * K2T * TILE_H) * sizeof(_Float16);  // 184320 B
  fused_mlp<<<B_TOTAL / 128, 256, lds, stream>>>(ws, b1, b2, b3, out);
}
